// NysAttention_26714696581586
// MI455X (gfx1250) — matrix operand order repaired and checked by reading, not compiled, not run on hardware
//
#include <hip/hip_runtime.h>

// ---------------------------------------------------------------------------
// Nystrom attention forward for MI455X (gfx1250, wave32, WMMA bf16->f32).
// Shapes: B=8, N=8192, C=384, D=128, L=64, H=1.
// GEMM operands pre-converted to bf16; staging via async global->LDS copies
// (double buffered, ASYNCcnt); WMMA f32 accumulate.
// ---------------------------------------------------------------------------

typedef __attribute__((ext_vector_type(16))) __bf16          v16bf;
typedef __attribute__((ext_vector_type(2)))  __bf16          v2bf;
typedef __attribute__((ext_vector_type(8)))  float           v8f;
typedef __attribute__((ext_vector_type(8)))  unsigned short  v8u16;
typedef __attribute__((ext_vector_type(16))) unsigned short  v16u16;

__device__ __forceinline__ unsigned short f2bf(float f) {
  __bf16 h = (__bf16)f;
  return __builtin_bit_cast(unsigned short, h);
}

__device__ __forceinline__ unsigned int f2bf2(float lo, float hi) {
  v2bf v;
  v[0] = (__bf16)lo;
  v[1] = (__bf16)hi;
  return __builtin_bit_cast(unsigned int, v);
}

// Async 16-byte global -> LDS copy (per-lane addresses), tracked by ASYNCcnt.
__device__ __forceinline__ void async_copy_b128(const void* lds_ptr, const void* gptr) {
  unsigned lds_off = (unsigned)(unsigned long long)lds_ptr;  // low 32 bits = LDS offset
  asm volatile("global_load_async_to_lds_b128 %0, %1, off"
               :: "v"(lds_off), "v"(gptr)
               : "memory");
}

// ---------------------------------------------------------------------------
// Bulk f32 -> bf16 conversion, 8 elements per thread. n8 = n/8.
// ---------------------------------------------------------------------------
__global__ __launch_bounds__(256)
void f32_to_bf16_kernel(const float* __restrict__ src,
                        unsigned short* __restrict__ dst, long long n8)
{
  long long i = (long long)blockIdx.x * 256 + threadIdx.x;
  if (i >= n8) return;
  const float4 a = reinterpret_cast<const float4*>(src)[i * 2];
  const float4 b = reinterpret_cast<const float4*>(src)[i * 2 + 1];
  uint4 o;
  o.x = f2bf2(a.x, a.y);
  o.y = f2bf2(a.z, a.w);
  o.z = f2bf2(b.x, b.y);
  o.w = f2bf2(b.z, b.w);
  reinterpret_cast<uint4*>(dst)[i] = o;
}

// ---------------------------------------------------------------------------
// Generic batched bf16 GEMM:  C[b] = alpha(n) * A[b] @ op(B[b]) (+bias+addend)
//   A: (M x K) bf16 row-major.
//   BT=1: op(B)=B^T, B stored (N x K) bf16;  BT=0: op(B)=B, stored (K x N).
//   Outputs: optional f32 (Cf) and/or bf16 (Cb) copies.
// Block: 128 threads = 4 waves; tile 64(M) x 16(N); K stepped by 64;
// double-buffered async global->LDS staging; 2 WMMAs / step, independent accs.
// Requires: M%64==0, N%16==0, K%64==0, lda/ldb(BT) % 8 == 0.
// ---------------------------------------------------------------------------
template <int BT>
__global__ __launch_bounds__(128)
void gemm_bf16_wmma_kernel(const unsigned short* __restrict__ A, int lda, long long sA,
                           const unsigned short* __restrict__ Bw, int ldb, long long sB,
                           int M, int N, int K,
                           float* __restrict__ Cf, int ldc, long long sC,
                           unsigned short* __restrict__ Cb, int ldcb, long long sCb,
                           float alpha0, int alpha_ncut,
                           const float* __restrict__ bias,
                           const float* __restrict__ addend, int ldadd, long long sAdd)
{
  __shared__ unsigned short Alds[2][64][72];   // 64 x 64 bf16, padded row stride
  __shared__ unsigned short Blds[2][16][72];   // Blds[buf][n][k] = op(B)[k][n]

  const int bz = blockIdx.z;
  A  += sA * bz;
  Bw += sB * bz;
  if (Cf)     Cf     += sC   * bz;
  if (Cb)     Cb     += sCb  * bz;
  if (addend) addend += sAdd * bz;

  const int m0   = blockIdx.x * 64;
  const int n0   = blockIdx.y * 16;
  const int t    = threadIdx.x;
  const int wave = t >> 5;
  const int lane = t & 31;
  const int half = lane >> 4;
  const int lrow = lane & 15;

  // staging index precompute
  const int arow = t >> 3;            // A: rows handled at f = t + 128*i
  const int ac8  = t & 7;
  const int brow = t >> 3;            // B: n within tile (0..15)
  const int bc8  = t & 7;             // B: k group of 8

  v8f acc0 = {0.f, 0.f, 0.f, 0.f, 0.f, 0.f, 0.f, 0.f};
  v8f acc1 = {0.f, 0.f, 0.f, 0.f, 0.f, 0.f, 0.f, 0.f};

  const int niter = K >> 6;

  // ---- stage one 64-wide K panel into buffer `buf`
  auto stage = [&](int buf, int k0) {
#pragma unroll
    for (int i = 0; i < 4; ++i) {
      int row = arow + 16 * i;        // f = t + 128*i -> row = f>>3
      async_copy_b128(&Alds[buf][row][ac8 * 8],
                      &A[(long long)(m0 + row) * lda + k0 + ac8 * 8]);
    }
    if (BT) {
      async_copy_b128(&Blds[buf][brow][bc8 * 8],
                      &Bw[(long long)(n0 + brow) * ldb + k0 + bc8 * 8]);
    } else {
      long long base = (long long)(k0 + bc8 * 8) * ldb + n0 + brow;
      v8u16 v;
#pragma unroll
      for (int j = 0; j < 8; ++j) v[j] = Bw[base + (long long)j * ldb];
      *reinterpret_cast<v8u16*>(&Blds[buf][brow][bc8 * 8]) = v;   // regular ds_store
    }
  };

  stage(0, 0);

  for (int i = 0; i < niter; ++i) {
    const int cur = i & 1;
    const bool pre = (i + 1 < niter);
    if (pre) stage(cur ^ 1, (i + 1) * 64);

    // wait for the *current* panel's async copies (next panel may stay in flight)
    if (pre) {
      if (BT) asm volatile("s_wait_asynccnt 0x5" ::: "memory");
      else    asm volatile("s_wait_asynccnt 0x4" ::: "memory");
    } else {
      asm volatile("s_wait_asynccnt 0x0" ::: "memory");
    }
    __syncthreads();

    // ---- two independent 16x16x32 WMMAs (k sub-chunks u=0,1)
#pragma unroll
    for (int u = 0; u < 2; ++u) {
      // A lane layout (16-bit A 16x32): lane<16 -> K {0..7,16..23},
      //                                 lane>=16 -> K {8..15,24..31}; row = lane%16
      const unsigned short* ap = &Alds[cur][wave * 16 + lrow][u * 32];
      v8u16 alo = *reinterpret_cast<const v8u16*>(ap + half * 8);
      v8u16 ahi = *reinterpret_cast<const v8u16*>(ap + 16 + half * 8);
      v16u16 au;
#pragma unroll
      for (int q = 0; q < 8; ++q) { au[q] = alo[q]; au[8 + q] = ahi[q]; }
      // B lane layout: lane<16 -> K {0..7,16..23} of column lane%16; lane>=16 -> K {8..15,24..31}
      const unsigned short* bp = &Blds[cur][lrow][u * 32];
      v8u16 blo = *reinterpret_cast<const v8u16*>(bp + half * 8);
      v8u16 bhi = *reinterpret_cast<const v8u16*>(bp + 16 + half * 8);
      v16u16 bu;
#pragma unroll
      for (int q = 0; q < 8; ++q) { bu[q] = blo[q]; bu[8 + q] = bhi[q]; }

      v16bf av = __builtin_bit_cast(v16bf, au);
      v16bf bv = __builtin_bit_cast(v16bf, bu);
      if (u == 0) {
        acc0 = __builtin_amdgcn_wmma_f32_16x16x32_bf16(false, av, false, bv,
                                                       (short)0, acc0, false, false);
      } else {
        acc1 = __builtin_amdgcn_wmma_f32_16x16x32_bf16(false, av, false, bv,
                                                       (short)0, acc1, false, false);
      }
    }
    __syncthreads();   // all waves done reading `cur` before it is restaged
  }

  v8f accv = acc0 + acc1;

  // ---- epilogue: C/D layout: VGPR j -> M = j + 8*half, N = lane%16
  const float alpha = (n0 < alpha_ncut) ? alpha0 : 1.0f;
  const int   gcol  = n0 + lrow;
  const float badd  = bias ? bias[gcol] : 0.0f;
#pragma unroll
  for (int j = 0; j < 8; ++j) {
    int grow  = m0 + wave * 16 + half * 8 + j;
    float val = alpha * accv[j] + badd;
    if (addend) val += addend[(long long)grow * ldadd + gcol];
    if (Cf) Cf[(long long)grow * ldc  + gcol] = val;
    if (Cb) Cb[(long long)grow * ldcb + gcol] = f2bf(val);
  }
}

// ---------------------------------------------------------------------------
// Row softmax, row length 64. One wave per row. Writes f32 and/or bf16.
// ---------------------------------------------------------------------------
__global__ __launch_bounds__(256)
void softmax64_kernel(const float* __restrict__ X, long long nrows,
                      unsigned short* __restrict__ Ob, float* __restrict__ Of)
{
  long long row = (long long)blockIdx.x * 8 + (threadIdx.x >> 5);
  if (row >= nrows) return;
  int lane = threadIdx.x & 31;
  const float* p = X + row * 64;
  float a = p[lane];
  float b = p[lane + 32];
  float m = fmaxf(a, b);
#pragma unroll
  for (int o = 16; o > 0; o >>= 1) m = fmaxf(m, __shfl_xor(m, o, 32));
  float ea = __expf(a - m);
  float eb = __expf(b - m);
  float s = ea + eb;
#pragma unroll
  for (int o = 16; o > 0; o >>= 1) s += __shfl_xor(s, o, 32);
  float inv = 1.0f / s;
  ea *= inv; eb *= inv;
  if (Of) { Of[row * 64 + lane] = ea; Of[row * 64 + lane + 32] = eb; }
  if (Ob) { Ob[row * 64 + lane] = f2bf(ea); Ob[row * 64 + lane + 32] = f2bf(eb); }
}

// ---------------------------------------------------------------------------
// Row softmax, long rows (L % 256 == 0). One block per row. bf16 output.
// ---------------------------------------------------------------------------
__global__ __launch_bounds__(256)
void softmax_long_kernel(const float* __restrict__ X, int L,
                         unsigned short* __restrict__ Ob)
{
  const float* p = X + (long long)blockIdx.x * L;
  unsigned short* o = Ob + (long long)blockIdx.x * L;
  int t = threadIdx.x;
  __shared__ float red[256];

  float m = -3.402823466e+38f;
  for (int i = t; i < L; i += 256) m = fmaxf(m, p[i]);
  red[t] = m; __syncthreads();
  for (int s = 128; s > 0; s >>= 1) {
    if (t < s) red[t] = fmaxf(red[t], red[t + s]);
    __syncthreads();
  }
  m = red[0]; __syncthreads();

  float sum = 0.0f;
  for (int i = t; i < L; i += 256) sum += __expf(p[i] - m);
  red[t] = sum; __syncthreads();
  for (int s = 128; s > 0; s >>= 1) {
    if (t < s) red[t] += red[t + s];
    __syncthreads();
  }
  float inv = 1.0f / red[0];
  for (int i = t; i < L; i += 256) o[i] = f2bf(__expf(p[i] - m) * inv);
}

// ---------------------------------------------------------------------------
// Landmark means: dst[(b*64+l)][d] = mean of 128 rows of src (f32, ld=384).
// grid = B*64 blocks, 128 threads. bf16 output.
// ---------------------------------------------------------------------------
__global__ __launch_bounds__(128)
void landmark_mean_kernel(const float* __restrict__ src, unsigned short* __restrict__ dst)
{
  int bl = blockIdx.x;              // b*64 + l
  int d  = threadIdx.x;
  int b  = bl >> 6;
  int l  = bl & 63;
  const float* p = src + ((long long)(b * 8192 + l * 128)) * 384 + d;
  float s = 0.0f;
  for (int i = 0; i < 128; ++i) s += p[(long long)i * 384];
  dst[(long long)bl * 128 + d] = f2bf(s * (1.0f / 128.0f));
}

// ---------------------------------------------------------------------------
// Newton-Schulz pseudo-inverse of 64x64 kernel, f32 in LDS; bf16 output.
// grid = B blocks, 256 threads.
// ---------------------------------------------------------------------------
__global__ __launch_bounds__(256)
void newton_pinv_kernel(const float* __restrict__ K2, unsigned short* __restrict__ Pb)
{
  const int b = blockIdx.x;
  const float* Kg = K2 + (long long)b * 4096;
  unsigned short* Pg = Pb + (long long)b * 4096;

  __shared__ float Vs[64][64];
  __shared__ float Xs[64][64];
  __shared__ float Ys[64][64];
  __shared__ float red[64];

  const int t = threadIdx.x;

  if (t < 64) {
    float s = 0.0f;
    for (int i = 0; i < 64; ++i) s += Kg[i * 64 + t];
    red[t] = s;
  }
  __syncthreads();
  if (t == 0) {
    float m = red[0];
    for (int j = 1; j < 64; ++j) m = fmaxf(m, red[j]);
    red[0] = 1.0f / m;
  }
  __syncthreads();
  const float invden = red[0];

  for (int e = t; e < 4096; e += 256) {
    int i = e >> 6, j = e & 63;
    Vs[i][j] = Kg[j * 64 + i] * invden;   // V = K^T / denom
  }
  __syncthreads();

  const int r  = t >> 2;
  const int c0 = (t & 3) * 16;
  float acc[16];

  for (int it = 0; it < 6; ++it) {
    // X = K @ V
#pragma unroll
    for (int j = 0; j < 16; ++j) acc[j] = 0.0f;
    for (int k = 0; k < 64; ++k) {
      float a = Kg[r * 64 + k];
#pragma unroll
      for (int j = 0; j < 16; ++j) acc[j] += a * Vs[k][c0 + j];
    }
#pragma unroll
    for (int j = 0; j < 16; ++j) Xs[r][c0 + j] = acc[j];
    __syncthreads();

    // Y = X @ (7I - X) = 7X - X@X
#pragma unroll
    for (int j = 0; j < 16; ++j) acc[j] = 7.0f * Xs[r][c0 + j];
    for (int k = 0; k < 64; ++k) {
      float a = Xs[r][k];
#pragma unroll
      for (int j = 0; j < 16; ++j) acc[j] -= a * Xs[k][c0 + j];
    }
#pragma unroll
    for (int j = 0; j < 16; ++j) Ys[r][c0 + j] = acc[j];
    __syncthreads();

    // X <- X @ (15I - Y) = 15X - X@Y  (in place via registers)
#pragma unroll
    for (int j = 0; j < 16; ++j) acc[j] = 15.0f * Xs[r][c0 + j];
    for (int k = 0; k < 64; ++k) {
      float a = Xs[r][k];
#pragma unroll
      for (int j = 0; j < 16; ++j) acc[j] -= a * Ys[k][c0 + j];
    }
    __syncthreads();
#pragma unroll
    for (int j = 0; j < 16; ++j) Xs[r][c0 + j] = acc[j];
    __syncthreads();

    // V <- 0.25 * (13V - V@X)  (in place via registers)
#pragma unroll
    for (int j = 0; j < 16; ++j) acc[j] = 13.0f * Vs[r][c0 + j];
    for (int k = 0; k < 64; ++k) {
      float a = Vs[r][k];
#pragma unroll
      for (int j = 0; j < 16; ++j) acc[j] -= a * Xs[k][c0 + j];
    }
    __syncthreads();
#pragma unroll
    for (int j = 0; j < 16; ++j) Vs[r][c0 + j] = 0.25f * acc[j];
    __syncthreads();
  }

  for (int e = t; e < 4096; e += 256) {
    int i = e >> 6, j = e & 63;
    Pg[e] = f2bf(Vs[i][j]);
  }
}

// ---------------------------------------------------------------------------
// Host-side orchestration
// ---------------------------------------------------------------------------
extern "C" void kernel_launch(void* const* d_in, const int* in_sizes, int n_in,
                              void* d_out, int out_size, void* d_ws, size_t ws_size,
                              hipStream_t stream)
{
  (void)in_sizes; (void)n_in; (void)out_size; (void)ws_size;

  const float* x      = (const float*)d_in[0];   // (8, 8192, 384)
  const float* qkv_w  = (const float*)d_in[1];   // (384, 384)
  const float* proj_w = (const float*)d_in[2];   // (128, 128)
  const float* proj_b = (const float*)d_in[3];   // (128,)
  float*       out    = (float*)d_out;           // (8, 8192, 128)

  constexpr long long Bn = 8, Nn = 8192, Ct = 384, Dd = 128, Ll = 64;
  constexpr long long MN = Bn * Nn;              // 65536
  const float INV_SCALE = 0.051031036307982884f; // 1/sqrt(384)

  // ---- workspace layout
  float* qkvf   = (float*)d_ws;                          // MN x 384
  float* logits = qkvf + (size_t)MN * Ct;                // MN x 64 (k1 / k3 reuse)
  float* k2f    = logits + (size_t)MN * Ll;              // B*64 x 64
  unsigned short* xb     = (unsigned short*)(k2f + (size_t)Bn * Ll * Ll);
  unsigned short* qkvwb  = xb     + (size_t)MN * Ct;     // 384 x 384
  unsigned short* projwb = qkvwb  + (size_t)Ct * Ct;     // 128 x 128
  unsigned short* qkvb   = projwb + (size_t)Dd * Dd;     // MN x 384
  unsigned short* qlmb   = qkvb   + (size_t)MN * Ct;     // B*64 x 128
  unsigned short* klmb   = qlmb   + (size_t)Bn * Ll * Dd;
  unsigned short* k1b    = klmb   + (size_t)Bn * Ll * Dd;  // MN x 64
  unsigned short* k3b    = k1b    + (size_t)MN * Ll;       // B*64 x 8192
  unsigned short* pinvb  = k3b    + (size_t)Bn * Ll * Nn;  // B*64 x 64
  unsigned short* k3vb   = pinvb  + (size_t)Bn * Ll * Ll;  // B*64 x 128
  unsigned short* midb   = k3vb   + (size_t)Bn * Ll * Dd;
  unsigned short* attnb  = midb   + (size_t)Bn * Ll * Dd;  // MN x 128

  // 0) pre-convert GEMM inputs to bf16
  f32_to_bf16_kernel<<<(MN * Ct) / (8 * 256), 256, 0, stream>>>(x, xb, (MN * Ct) / 8);
  f32_to_bf16_kernel<<<(Ct * Ct) / (8 * 256), 256, 0, stream>>>(qkv_w, qkvwb, (Ct * Ct) / 8);
  f32_to_bf16_kernel<<<(Dd * Dd) / (8 * 256), 256, 0, stream>>>(proj_w, projwb, (Dd * Dd) / 8);

  // 1) qkv = x @ qkv_w^T ; q columns [0,128) scaled by 1/sqrt(384); f32+bf16 out
  gemm_bf16_wmma_kernel<1><<<dim3(MN / 64, Ct / 16, 1), 128, 0, stream>>>(
      xb, Ct, 0, qkvwb, Ct, 0, (int)MN, (int)Ct, (int)Ct,
      qkvf, Ct, 0, qkvb, Ct, 0, INV_SCALE, /*ncut=*/128, nullptr, nullptr, 0, 0);

  // 2) landmark means of q and k (f32 in, bf16 out)
  landmark_mean_kernel<<<Bn * Ll, 128, 0, stream>>>(qkvf + 0,   qlmb);
  landmark_mean_kernel<<<Bn * Ll, 128, 0, stream>>>(qkvf + 128, klmb);

  // 3) kernel_1 = softmax(q @ klm^T) -> bf16
  gemm_bf16_wmma_kernel<1><<<dim3(Nn / 64, Ll / 16, Bn), 128, 0, stream>>>(
      qkvb + 0, Ct, Nn * Ct, klmb, Dd, Ll * Dd, (int)Nn, (int)Ll, (int)Dd,
      logits, Ll, Nn * Ll, nullptr, 0, 0, 1.0f, 0, nullptr, nullptr, 0, 0);
  softmax64_kernel<<<MN / 8, 256, 0, stream>>>(logits, MN, k1b, nullptr);

  // 4) kernel_2 = softmax(qlm @ klm^T) -> f32 (feeds Newton)
  gemm_bf16_wmma_kernel<1><<<dim3(1, Ll / 16, Bn), 128, 0, stream>>>(
      qlmb, Dd, Ll * Dd, klmb, Dd, Ll * Dd, (int)Ll, (int)Ll, (int)Dd,
      k2f, Ll, Ll * Ll, nullptr, 0, 0, 1.0f, 0, nullptr, nullptr, 0, 0);
  softmax64_kernel<<<(Bn * Ll) / 8, 256, 0, stream>>>(k2f, Bn * Ll, nullptr, k2f);

  // 5) kernel_3 = softmax(qlm @ k^T) -> bf16
  gemm_bf16_wmma_kernel<1><<<dim3(1, Nn / 16, Bn), 128, 0, stream>>>(
      qlmb, Dd, Ll * Dd, qkvb + 128, Ct, Nn * Ct, (int)Ll, (int)Nn, (int)Dd,
      logits, Nn, Ll * Nn, nullptr, 0, 0, 1.0f, 0, nullptr, nullptr, 0, 0);
  softmax_long_kernel<<<Bn * Ll, 256, 0, stream>>>(logits, (int)Nn, k3b);

  // 6) pinv = iterative_inv(kernel_2) -> bf16
  newton_pinv_kernel<<<Bn, 256, 0, stream>>>(k2f, pinvb);

  // 7) k3v = kernel_3 @ v -> bf16
  gemm_bf16_wmma_kernel<0><<<dim3(1, Dd / 16, Bn), 128, 0, stream>>>(
      k3b, Nn, Ll * Nn, qkvb + 256, Ct, Nn * Ct, (int)Ll, (int)Dd, (int)Nn,
      nullptr, 0, 0, k3vb, Dd, Ll * Dd, 1.0f, 0, nullptr, nullptr, 0, 0);

  // 8) mid = pinv @ k3v -> bf16
  gemm_bf16_wmma_kernel<0><<<dim3(1, Dd / 16, Bn), 128, 0, stream>>>(
      pinvb, Ll, Ll * Ll, k3vb, Dd, Ll * Dd, (int)Ll, (int)Dd, (int)Ll,
      nullptr, 0, 0, midb, Dd, Ll * Dd, 1.0f, 0, nullptr, nullptr, 0, 0);

  // 9) attn = kernel_1 @ mid -> bf16
  gemm_bf16_wmma_kernel<0><<<dim3(Nn / 64, Dd / 16, Bn), 128, 0, stream>>>(
      k1b, Ll, Nn * Ll, midb, Dd, Ll * Dd, (int)Nn, (int)Dd, (int)Ll,
      nullptr, 0, 0, attnb, Dd, Nn * Dd, 1.0f, 0, nullptr, nullptr, 0, 0);

  // 10) out = attn @ proj_w^T + proj_b + v  (batch folded; v = qkvf col 256+)
  gemm_bf16_wmma_kernel<1><<<dim3(MN / 64, Dd / 16, 1), 128, 0, stream>>>(
      attnb, Dd, 0, projwb, Dd, 0, (int)MN, (int)Dd, (int)Dd,
      out, Dd, 0, nullptr, 0, 0, 1.0f, 0,
      proj_b, /*addend=*/qkvf + 256, /*ldadd=*/(int)Ct, 0);
}
